// Seq2SeqGRU_25658134627074
// MI455X (gfx1250) — matrix operand order repaired and checked by reading, not compiled, not run on hardware
//
#include <hip/hip_runtime.h>
#include <hip/hip_bf16.h>

// ---------------------------------------------------------------------------
// Seq2Seq GRU (B=256, S=512, F=128, H=512, C=100) for gfx1250 / MI455X.
// Recurrent GEMMs use v_wmma_f32_16x16x32_bf16 (wave32, 16x16 tiles).
// x is pre-converted to bf16 (one streaming pass) so the 512 sequential
// encoder steps run pure b128-load + WMMA, no VALU conversion in the loop.
// ---------------------------------------------------------------------------

#define Bsz 256
#define Ssz 512
#define Fsz 128
#define Hsz 512
#define Csz 100

typedef __attribute__((ext_vector_type(16))) __bf16 v16bf;
typedef __attribute__((ext_vector_type(8)))  float  v8f;

union BF16x16 {
  unsigned short s[16];
  uint4          u[2];
  v16bf          v;
};

static __device__ __forceinline__ unsigned short f2bfu(float f) {
  union { float f; unsigned u; } a; a.f = f;
  unsigned u = a.u;
  return (unsigned short)((u + 0x7FFFu + ((u >> 16) & 1u)) >> 16);
}

// A-matrix (16x32 bf16, MxK): lane L holds row M = L%16.
// L<16  -> K in {0..7} u {16..23};  L>=16 -> K in {8..15} u {24..31}.
static __device__ __forceinline__ v16bf load_a_bf(const unsigned short* A, long ld,
                                                  int row, int kBase, bool hi) {
  const unsigned short* p = A + (long)row * ld + kBase + (hi ? 8 : 0);
  BF16x16 r;
  r.u[0] = *(const uint4*)(p);
  r.u[1] = *(const uint4*)(p + 16);
  return r.v;
}

// Same A layout but converting f32 source to bf16 on the fly (fallback path).
static __device__ __forceinline__ v16bf load_a_f32(const float* A, long ld,
                                                   int row, int kBase, bool hi) {
  const float* p = A + (long)row * ld + kBase + (hi ? 8 : 0);
  BF16x16 r;
#pragma unroll
  for (int i = 0; i < 8; ++i) {
    r.s[i]     = f2bfu(p[i]);
    r.s[8 + i] = f2bfu(p[16 + i]);
  }
  return r.v;
}

// B-matrix (32x16 bf16, KxN), B[k][n] = W[gRowBase+n][kBase+k] with W row-major:
// lane L holds column n = L%16; L<16 -> K in {0..7} u {16..23}, L>=16 -> K in {8..15} u {24..31}.
static __device__ __forceinline__ v16bf load_b_bf(const unsigned short* W, long ldw,
                                                  int gRow, int kBase, bool hi) {
  const unsigned short* p = W + (long)gRow * ldw + kBase + (hi ? 8 : 0);
  BF16x16 r;
  r.u[0] = *(const uint4*)(p);
  r.u[1] = *(const uint4*)(p + 16);
  return r.v;
}

static __device__ __forceinline__ v8f wmma_bf16(v16bf a, v16bf b, v8f c) {
  return __builtin_amdgcn_wmma_f32_16x16x32_bf16(false, a, false, b,
                                                 (short)0, c, false, false);
}

// ---------------------------------------------------------------------------
// One GRU step: h_new = GRUCell(h_old, input).
// Encoder mode: xbf_t (bf16, preferred) or x_t (f32 fallback) non-null.
// Decoder mode: dec_in non-null (scalar input, f32 dec_Wih [3H,1]).
// Grid: 64 blocks x 256 threads = 512 waves; one wave per 16x16 output tile.
// ---------------------------------------------------------------------------
__global__ __launch_bounds__(256)
void gru_step_kernel(const unsigned short* __restrict__ h_bf,   // [B,H] bf16
                     const float*          __restrict__ h_f32,  // [B,H] f32
                     const unsigned short* __restrict__ Whh_bf, // [3H,H] bf16
                     const float*          __restrict__ bih,    // [3H]
                     const float*          __restrict__ bhh,    // [3H]
                     const unsigned short* __restrict__ xbf_t,  // bf16 or null
                     const float*          __restrict__ x_t,    // f32 or null
                     long x_ld,
                     const unsigned short* __restrict__ Wih_bf, // [3H,F] bf16 or null
                     const float*          __restrict__ dec_in,  // [B] or null
                     const float*          __restrict__ dec_Wih, // [3H,1] f32 or null
                     float*                __restrict__ h_out_f32,
                     unsigned short*       __restrict__ h_out_bf) {
  const int wid  = (blockIdx.x * blockDim.x + threadIdx.x) >> 5;
  const int lane = threadIdx.x & 31;
  const int tileRow = wid >> 5;        // 0..15  (batch tiles)
  const int tileCol = wid & 31;        // 0..31  (hidden-col tiles)
  const int rowBase = tileRow * 16;
  const int colBase = tileCol * 16;
  const int m  = lane & 15;
  const bool hi = lane >= 16;
  const int j  = colBase + m;          // this lane's output column

  v8f acc_r  = {};   // r gate: input + hidden parts
  v8f acc_z  = {};   // z gate: input + hidden parts
  v8f acc_ni = {};   // n gate: input part only
  v8f acc_nh = {};   // n gate: hidden part only

  // ---- recurrent GEMM: h @ Whh^T, K = H ----
  for (int k = 0; k < Hsz; k += 32) {
    v16bf a  = load_a_bf(h_bf, Hsz, rowBase + m, k, hi);
    v16bf br = load_b_bf(Whh_bf, Hsz, 0 * Hsz + colBase + m, k, hi);
    v16bf bz = load_b_bf(Whh_bf, Hsz, 1 * Hsz + colBase + m, k, hi);
    v16bf bn = load_b_bf(Whh_bf, Hsz, 2 * Hsz + colBase + m, k, hi);
    acc_r  = wmma_bf16(a, br, acc_r);
    acc_z  = wmma_bf16(a, bz, acc_z);
    acc_nh = wmma_bf16(a, bn, acc_nh);
  }

  // ---- encoder input GEMM: x_t @ Wih^T, K = F ----
  if (xbf_t != nullptr) {
    for (int k = 0; k < Fsz; k += 32) {
      v16bf a  = load_a_bf(xbf_t, x_ld, rowBase + m, k, hi);
      v16bf br = load_b_bf(Wih_bf, Fsz, 0 * Hsz + colBase + m, k, hi);
      v16bf bz = load_b_bf(Wih_bf, Fsz, 1 * Hsz + colBase + m, k, hi);
      v16bf bn = load_b_bf(Wih_bf, Fsz, 2 * Hsz + colBase + m, k, hi);
      acc_r  = wmma_bf16(a, br, acc_r);
      acc_z  = wmma_bf16(a, bz, acc_z);
      acc_ni = wmma_bf16(a, bn, acc_ni);
    }
  } else if (x_t != nullptr) {
    for (int k = 0; k < Fsz; k += 32) {
      v16bf a  = load_a_f32(x_t, x_ld, rowBase + m, k, hi);
      v16bf br = load_b_bf(Wih_bf, Fsz, 0 * Hsz + colBase + m, k, hi);
      v16bf bz = load_b_bf(Wih_bf, Fsz, 1 * Hsz + colBase + m, k, hi);
      v16bf bn = load_b_bf(Wih_bf, Fsz, 2 * Hsz + colBase + m, k, hi);
      acc_r  = wmma_bf16(a, br, acc_r);
      acc_z  = wmma_bf16(a, bz, acc_z);
      acc_ni = wmma_bf16(a, bn, acc_ni);
    }
  }

  // ---- biases (per output column) ----
  const float br_  = bih[j] + bhh[j];
  const float bz_  = bih[Hsz + j] + bhh[Hsz + j];
  const float bin_ = bih[2 * Hsz + j];
  const float bhn_ = bhh[2 * Hsz + j];
  float wr = 0.f, wz = 0.f, wn = 0.f;
  if (dec_in != nullptr) {
    wr = dec_Wih[j];
    wz = dec_Wih[Hsz + j];
    wn = dec_Wih[2 * Hsz + j];
  }

  // ---- GRU nonlinearity + blend (C/D layout: VGPR i, lane -> M = i + 8*hi) ----
#pragma unroll
  for (int i = 0; i < 8; ++i) {
    const int b = rowBase + i + (hi ? 8 : 0);
    const float hold = h_f32[(long)b * Hsz + j];
    float ir_hr = acc_r[i] + br_;
    float iz_hz = acc_z[i] + bz_;
    float i_n   = acc_ni[i] + bin_;
    if (dec_in != nullptr) {
      const float d = dec_in[b];
      ir_hr += d * wr;
      iz_hz += d * wz;
      i_n   += d * wn;
    }
    const float r  = 1.f / (1.f + __expf(-ir_hr));
    const float z  = 1.f / (1.f + __expf(-iz_hz));
    const float nn = tanhf(i_n + r * (acc_nh[i] + bhn_));
    const float hn = (1.f - z) * nn + z * hold;
    h_out_f32[(long)b * Hsz + j] = hn;
    h_out_bf [(long)b * Hsz + j] = f2bfu(hn);
  }
}

// ---------------------------------------------------------------------------
// Decoder head: logits = tanh(h2) @ W^T + b, softmax -> log_softmax(prob) CE,
// argmax feedback. One block per batch row.
// ---------------------------------------------------------------------------
__global__ __launch_bounds__(128)
void head_kernel(const float* __restrict__ h2,     // [B,H]
                 const float* __restrict__ W,      // [C,H]
                 const float* __restrict__ bias,   // [C]
                 const int*   __restrict__ y,      // [B,C]
                 int step,
                 float* __restrict__ preds,        // [B,C] (d_out)
                 float* __restrict__ dec_in_next,  // [B]
                 float* __restrict__ loss_acc,     // scalar (d_out + B*C)
                 float scale) {
  __shared__ float th[Hsz];
  __shared__ float logits[128];
  const int b = blockIdx.x;
  const int t = threadIdx.x;

  for (int k = t; k < Hsz; k += blockDim.x)
    th[k] = tanhf(h2[(long)b * Hsz + k]);
  __syncthreads();

  if (t < Csz) {
    float acc = bias[t];
    const float* wr = W + (long)t * Hsz;
    for (int k = 0; k < Hsz; ++k) acc += th[k] * wr[k];
    logits[t] = acc;
  }
  __syncthreads();

  if (t == 0) {
    // softmax (stable)
    float mx = -1e30f; int amax = 0;
    for (int c = 0; c < Csz; ++c)
      if (logits[c] > mx) { mx = logits[c]; amax = c; }
    float se = 0.f;
    for (int c = 0; c < Csz; ++c) se += __expf(logits[c] - mx);
    // log_softmax of prob: logp[c] = prob[c] - logsumexp(prob); max(prob)=1/se.
    const float m2 = 1.f / se;
    float se2 = 0.f;
    for (int c = 0; c < Csz; ++c)
      se2 += __expf(__expf(logits[c] - mx) / se - m2);
    const float lse2 = m2 + __logf(se2);
    const int   yt   = y[b * Csz + step];
    const float pyt  = __expf(logits[yt] - mx) / se;
    const float logp = pyt - lse2;
    atomicAdd(loss_acc, -logp * scale);
    preds[(long)b * Csz + step] = (float)amax;
    dec_in_next[b] = (float)amax;
  }
}

// ---------------------------------------------------------------------------
// Packed f32 -> bf16 conversion: 2 elements per thread (one dword store).
__global__ void cvt_bf16_pk_kernel(const float2* __restrict__ src,
                                   unsigned int* __restrict__ dst, long n2) {
  long i = (long)blockIdx.x * blockDim.x + threadIdx.x;
  if (i < n2) {
    float2 v = src[i];
    dst[i] = (unsigned)f2bfu(v.x) | ((unsigned)f2bfu(v.y) << 16);
  }
}

__global__ void init_state_kernel(float* hf, unsigned short* hb,
                                  float* din, float* loss, int nh, int nb) {
  int i = blockIdx.x * blockDim.x + threadIdx.x;
  if (i < nh) { hf[i] = 0.f; hb[i] = 0; }
  if (i < nb) din[i] = 0.f;
  if (i == 0) loss[0] = 0.f;
}

// ---------------------------------------------------------------------------
extern "C" void kernel_launch(void* const* d_in, const int* in_sizes, int n_in,
                              void* d_out, int out_size, void* d_ws, size_t ws_size,
                              hipStream_t stream) {
  const float* x        = (const float*)d_in[0];   // [B,S,F]
  const int*   y        = (const int*)  d_in[1];   // [B,C]
  const float* enc_Wih  = (const float*)d_in[2];   // [3H,F]
  const float* enc_Whh  = (const float*)d_in[3];   // [3H,H]
  const float* enc_bih  = (const float*)d_in[4];
  const float* enc_bhh  = (const float*)d_in[5];
  const float* dec_Wih  = (const float*)d_in[6];   // [3H,1]
  const float* dec_Whh  = (const float*)d_in[7];   // [3H,H]
  const float* dec_bih  = (const float*)d_in[8];
  const float* dec_bhh  = (const float*)d_in[9];
  const float* dense_W  = (const float*)d_in[10];  // [C,H]
  const float* dense_b  = (const float*)d_in[11];

  // ---- workspace layout ----
  char* ws = (char*)d_ws;
  size_t used = 0;
  auto alloc = [&](size_t bytes) {
    char* p = ws;
    size_t padded = (bytes + 255) & ~(size_t)255;
    ws += padded;
    used += padded;
    return p;
  };
  float*          hf[2];
  unsigned short* hb[2];
  hf[0] = (float*)alloc((size_t)Bsz * Hsz * 4);
  hf[1] = (float*)alloc((size_t)Bsz * Hsz * 4);
  hb[0] = (unsigned short*)alloc((size_t)Bsz * Hsz * 2);
  hb[1] = (unsigned short*)alloc((size_t)Bsz * Hsz * 2);
  unsigned short* wih_bf  = (unsigned short*)alloc((size_t)3 * Hsz * Fsz * 2);
  unsigned short* whh_bf  = (unsigned short*)alloc((size_t)3 * Hsz * Hsz * 2);
  unsigned short* dwhh_bf = (unsigned short*)alloc((size_t)3 * Hsz * Hsz * 2);
  float* din[2];
  din[0] = (float*)alloc((size_t)Bsz * 4);
  din[1] = (float*)alloc((size_t)Bsz * 4);

  // Optional 32MB bf16 staging of x (preferred path); fall back if ws too small.
  const size_t x_elems = (size_t)Bsz * Ssz * Fsz;
  unsigned short* x_bf = nullptr;
  if (used + x_elems * 2 + 256 <= ws_size) {
    x_bf = (unsigned short*)alloc(x_elems * 2);
  }

  float* out_preds = (float*)d_out;          // [B,C]
  float* out_loss  = out_preds + Bsz * Csz;  // scalar

  // ---- weight (and optionally x) conversion f32 -> bf16, packed 2/thread ----
  const long n_wih2 = (long)3 * Hsz * Fsz / 2;
  const long n_whh2 = (long)3 * Hsz * Hsz / 2;
  cvt_bf16_pk_kernel<<<(int)((n_wih2 + 255) / 256), 256, 0, stream>>>(
      (const float2*)enc_Wih, (unsigned int*)wih_bf, n_wih2);
  cvt_bf16_pk_kernel<<<(int)((n_whh2 + 255) / 256), 256, 0, stream>>>(
      (const float2*)enc_Whh, (unsigned int*)whh_bf, n_whh2);
  cvt_bf16_pk_kernel<<<(int)((n_whh2 + 255) / 256), 256, 0, stream>>>(
      (const float2*)dec_Whh, (unsigned int*)dwhh_bf, n_whh2);
  if (x_bf) {
    const long n_x2 = (long)(x_elems / 2);
    cvt_bf16_pk_kernel<<<(int)((n_x2 + 255) / 256), 256, 0, stream>>>(
        (const float2*)x, (unsigned int*)x_bf, n_x2);
  }

  // ---- zero h0, dec_in0, loss ----
  init_state_kernel<<<(Bsz * Hsz + 255) / 256, 256, 0, stream>>>(
      hf[0], hb[0], din[0], out_loss, Bsz * Hsz, Bsz);

  // ---- encoder scan ----
  int cur = 0;
  for (int t = 0; t < Ssz; ++t) {
    int nxt = cur ^ 1;
    gru_step_kernel<<<64, 256, 0, stream>>>(
        hb[cur], hf[cur], whh_bf, enc_bih, enc_bhh,
        x_bf ? (x_bf + (long)t * Fsz) : nullptr,
        x_bf ? nullptr : (x + (long)t * Fsz),
        (long)Ssz * Fsz, wih_bf,
        nullptr, nullptr,
        hf[nxt], hb[nxt]);
    cur = nxt;
  }

  // ---- autoregressive decoder ----
  const float scale = 1.0f / ((float)Bsz * (float)Bsz * (float)Csz);
  int dcur = 0;
  for (int s = 0; s < Csz; ++s) {
    int nxt = cur ^ 1, dnxt = dcur ^ 1;
    gru_step_kernel<<<64, 256, 0, stream>>>(
        hb[cur], hf[cur], dwhh_bf, dec_bih, dec_bhh,
        nullptr, nullptr, 0, nullptr,
        din[dcur], dec_Wih,
        hf[nxt], hb[nxt]);
    head_kernel<<<Bsz, 128, 0, stream>>>(
        hf[nxt], dense_W, dense_b, y, s,
        out_preds, din[dnxt], out_loss, scale);
    cur = nxt; dcur = dnxt;
  }
  (void)in_sizes; (void)n_in; (void)out_size;
}
